// DILSTMGaus_45535243272843
// MI455X (gfx1250) — matrix operand order repaired and checked by reading, not compiled, not run on hardware
//
#include <hip/hip_runtime.h>
#include <hip/hip_bf16.h>
#include <math.h>

typedef __attribute__((ext_vector_type(16))) __bf16 v16bf;
typedef __attribute__((ext_vector_type(8)))  float  v8f;

#define T_STEPS 512
#define ZSTR    1204   // 8*1204 % 64 != 0 -> dodges LDS bank conflicts on z stores
#define NTHREADS 256

__device__ __forceinline__ unsigned short f2bf_bits(float f) {
    unsigned u = __builtin_bit_cast(unsigned, f);
    unsigned r = u + 0x7FFFu + ((u >> 16) & 1u);   // round-to-nearest-even
    return (unsigned short)(r >> 16);
}
__device__ __forceinline__ float hsig(float z) {
    return fminf(fmaxf(0.2f * z + 0.5f, 0.f), 1.f);
}

union ABfrag { v16bf v; unsigned u32[8]; uint4 q[2]; };

// ---------------------------------------------------------------------------
// Packed bf16 B-fragments: [ntile(77)][kchunk(11)][512 bf16].
// Tiles 0..74  : z-GEMM weights  [recurrent_kernel(300->320) ; kernel(25->32)]
// Tile 75      : MDN cols 0..15  = [wa(8) | wm(8)], K chunks 0..9 only
// Tile 76      : MDN cols 16..31 = [ws(8) | pad(8)]
// B (32x16 bf16) lane layout: lanes 0-15 K=0..7,16..23 (2/VGPR), lanes 16-31 K=8..15,24..31;
// N = lane & 15.  Each lane's 16 bf16 stored contiguously.
// ---------------------------------------------------------------------------
__global__ void pack_weights(const float* __restrict__ kern,   // (25,1200)
                             const float* __restrict__ rk,     // (300,1200)
                             const float* __restrict__ wa,     // (300,8)
                             const float* __restrict__ wm,     // (300,8)
                             const float* __restrict__ ws,     // (300,8)
                             unsigned short* __restrict__ packed) {
    int blk = blockIdx.x;           // 0..846
    int ntile = blk / 11, cc = blk % 11;
    for (int e = threadIdx.x; e < 512; e += blockDim.x) {
        int lane = e >> 4, idx = e & 15;
        int kloc = ((lane >> 4) * 8) + (idx & 7) + ((idx & 8) ? 16 : 0);   // 0..31 within chunk
        float v = 0.f;
        if (ntile < 75) {
            int n = ntile * 16 + (lane & 15);
            if (cc < 10) {                      // recurrent, K padded 300->320
                int kg = cc * 32 + kloc;
                if (kg < 300) v = rk[kg * 1200 + n];
            } else if (kloc < 25) {             // input/gate, K padded 25->32
                v = kern[kloc * 1200 + n];
            }
        } else {                                // MDN head tiles
            int col = (ntile - 75) * 16 + (lane & 15);   // 0..31
            if (cc < 10) {
                int kg = cc * 32 + kloc;
                if (kg < 300) {
                    if (col < 8)       v = wa[kg * 8 + col];
                    else if (col < 16) v = wm[kg * 8 + (col - 8)];
                    else if (col < 24) v = ws[kg * 8 + (col - 16)];
                }
            }
        }
        packed[(size_t)(ntile * 11 + cc) * 512 + e] = f2bf_bits(v);
    }
}

// ---------------------------------------------------------------------------
// Persistent scan kernel: block b owns batch rows [16b, 16b+16), loops t=0..511.
// 256 threads = 8 waves; waves split the 75 z-tiles; waves 0/1 run the MDN tiles.
// ---------------------------------------------------------------------------
__global__ void __launch_bounds__(NTHREADS)
dilstm_main(const float* __restrict__ x,      // (2048,512,25)
            const float* __restrict__ bias,   // (1200)
            const float* __restrict__ w1, const float* __restrict__ b1,
            const float* __restrict__ w2, const float* __restrict__ b2,
            const float* __restrict__ ba, const float* __restrict__ bm,
            const float* __restrict__ bs,
            const unsigned short* __restrict__ packedW,
            float* __restrict__ out) {
    __shared__ float zLds[16 * ZSTR];          // z gates; h aliased into zi slots
    __shared__ float cLds[16 * 304];           // cell state
    __shared__ unsigned short aBf[16 * 352];   // A bf16: [h(300) pad | gate(25) pad]
    __shared__ float prevLds[16 * 25];
    __shared__ float xLds[16 * 25];
    __shared__ float vinLds[16 * 52];
    __shared__ float g1Lds[16 * 52];
    __shared__ float mdnZ[16 * 32];            // cols: 0-7 alpha-z, 8-15 mu, 16-23 sigma-z
    __shared__ float cmbLds[16];
    __shared__ float biasLds[1200];
    __shared__ float mdnBias[32];
    __shared__ float w1Lds[50 * 52];
    __shared__ float b1Lds[52];
    __shared__ float w2Lds[50 * 24];
    __shared__ float b2Lds[24];

    const int tid  = threadIdx.x;
    const int lane = tid & 31;
    const int wave = tid >> 5;
    const int baseRow = blockIdx.x * 16;

    // ---- one-time init -------------------------------------------------
    for (int i = tid; i < 16 * 304; i += NTHREADS) cLds[i] = 0.f;
    for (int i = tid; i < 16 * 352; i += NTHREADS) aBf[i] = 0;       // h0 = 0
    for (int i = tid; i < 16 * 25;  i += NTHREADS) prevLds[i] = 0.f; // prev0 = 0
    for (int i = tid; i < 1200; i += NTHREADS) { biasLds[i] = bias[i]; w2Lds[i] = w2[i]; }
    for (int i = tid; i < 2500; i += NTHREADS) w1Lds[(i / 50) * 52 + (i % 50)] = w1[i];
    if (tid < 50) b1Lds[tid] = b1[tid];
    if (tid < 24) b2Lds[tid] = b2[tid];
    if (tid < 32) {
        float bv = 0.f;
        if (tid < 8)       bv = ba[tid];
        else if (tid < 16) bv = bm[tid - 8];
        else if (tid < 24) bv = bs[tid - 16];
        mdnBias[tid] = bv;
    }
    __syncthreads();

    // per-lane fragment geometry (16-bit A 16x32 layout, ISA 7.12.2)
    const int mA    = lane & 15;
    const int kHalf = (lane >> 4) * 8;
    const int nCol  = lane & 15;
    const int mBase = (lane >> 4) * 8;

    for (int t = 0; t < T_STEPS; ++t) {
        // ---- phase 0: stage x_t (+ prefetch x_{t+1}) -------------------
        for (int i = tid; i < 400; i += NTHREADS) {
            int m = i / 25, j = i % 25;
            size_t base = (size_t)(baseRow + m) * (T_STEPS * 25) + t * 25 + j;
            xLds[i] = x[base];
            if (t + 1 < T_STEPS) __builtin_prefetch(&x[base + 25], 0, 1);
        }
        __syncthreads();

        // ---- phase 1a: combined + MLP input vector ---------------------
        if (tid < 16) {
            int m = tid;
            float il = xLds[m * 25 + 24], pl = prevLds[m * 25 + 24];
            float cm = il + pl;
            cmbLds[m] = cm;
            aBf[m * 352 + 344] = f2bf_bits(cm);   // gate_out col 24 = combined
        }
        for (int i = tid; i < 800; i += NTHREADS) {
            int m = i / 50, k = i % 50;
            float il = xLds[m * 25 + 24], pl = prevLds[m * 25 + 24];
            float dn = fmaxf(il + pl, 1e-8f);
            float v;
            if (k < 24)       v = xLds[m * 25 + k];
            else if (k == 24) v = il / dn;
            else if (k < 49)  v = prevLds[m * 25 + (k - 25)];
            else              v = pl / dn;
            vinLds[m * 52 + k] = v;
        }
        __syncthreads();

        // ---- phase 1b: g1 = relu(vin @ w1 + b1) ------------------------
        for (int i = tid; i < 800; i += NTHREADS) {
            int m = i / 50, jo = i % 50;
            float s = b1Lds[jo];
            #pragma unroll 5
            for (int k = 0; k < 50; ++k) s += vinLds[m * 52 + k] * w1Lds[k * 52 + jo];
            g1Lds[m * 52 + jo] = fmaxf(s, 0.f);
        }
        __syncthreads();

        // ---- phase 1c: g2 = g1 @ w2 + b2 -> gate slots of A ------------
        for (int i = tid; i < 384; i += NTHREADS) {
            int m = i / 24, jo = i % 24;
            float s = b2Lds[jo];
            #pragma unroll 5
            for (int k = 0; k < 50; ++k) s += g1Lds[m * 52 + k] * w2Lds[k * 24 + jo];
            aBf[m * 352 + 320 + jo] = f2bf_bits(s);
        }
        __syncthreads();

        // ---- phase 2: z = A(16x352) @ Wpacked(352x1200) + bias ---------
        {
            ABfrag aF[11];
            #pragma unroll
            for (int cc = 0; cc < 11; ++cc) {
                #pragma unroll
                for (int v = 0; v < 8; ++v) {
                    int k0 = cc * 32 + kHalf + ((v & 4) ? 16 : 0) + ((v & 3) * 2);
                    aF[cc].u32[v] = *(const unsigned*)&aBf[mA * 352 + k0];
                }
            }
            for (int jt = wave; jt < 75; jt += 8) {
                float bv = biasLds[jt * 16 + nCol];
                v8f acc = {bv, bv, bv, bv, bv, bv, bv, bv};
                const unsigned short* bp = packedW + (size_t)jt * 11 * 512 + lane * 16;
                #pragma unroll
                for (int cc = 0; cc < 11; ++cc) {
                    ABfrag bF;
                    bF.q[0] = *(const uint4*)(bp + (size_t)cc * 512);
                    bF.q[1] = *(const uint4*)(bp + (size_t)cc * 512 + 8);
#if defined(__gfx1250__)
                    acc = __builtin_amdgcn_wmma_f32_16x16x32_bf16(
                        false, aF[cc].v, false, bF.v, (short)0, acc, false, false);
#endif
                }
                #pragma unroll
                for (int v = 0; v < 8; ++v)
                    zLds[(mBase + v) * ZSTR + jt * 16 + nCol] = acc[v];
            }
        }
        __syncthreads();

        // ---- phase 3: LSTM pointwise -----------------------------------
        for (int i = tid; i < 16 * 300; i += NTHREADS) {
            int m = i / 300, u = i % 300;
            float zi = zLds[m * ZSTR + u];
            float zf = zLds[m * ZSTR + 300 + u];
            float zc = zLds[m * ZSTR + 600 + u];
            float zo = zLds[m * ZSTR + 900 + u];
            float ig = hsig(zi), fg = hsig(zf), og = hsig(zo);
            float cn = fg * cLds[m * 304 + u] + ig * tanhf(zc);
            float hn = og * tanhf(cn);
            cLds[m * 304 + u] = cn;
            zLds[m * ZSTR + u] = hn;             // (kept for debug symmetry)
            aBf[m * 352 + u] = f2bf_bits(hn);    // h_new -> A matrix (bf16)
        }
        __syncthreads();

        // ---- phase 4: MDN head via WMMA (h_new @ [wa|wm|ws]) -----------
        if (wave < 2) {
            ABfrag aF[10];
            #pragma unroll
            for (int cc = 0; cc < 10; ++cc) {
                #pragma unroll
                for (int v = 0; v < 8; ++v) {
                    int k0 = cc * 32 + kHalf + ((v & 4) ? 16 : 0) + ((v & 3) * 2);
                    aF[cc].u32[v] = *(const unsigned*)&aBf[mA * 352 + k0];
                }
            }
            int jt = 75 + wave;
            int colBase = wave * 16;
            float bv = mdnBias[colBase + nCol];
            v8f acc = {bv, bv, bv, bv, bv, bv, bv, bv};
            const unsigned short* bp = packedW + (size_t)jt * 11 * 512 + lane * 16;
            #pragma unroll
            for (int cc = 0; cc < 10; ++cc) {
                ABfrag bF;
                bF.q[0] = *(const uint4*)(bp + (size_t)cc * 512);
                bF.q[1] = *(const uint4*)(bp + (size_t)cc * 512 + 8);
#if defined(__gfx1250__)
                acc = __builtin_amdgcn_wmma_f32_16x16x32_bf16(
                    false, aF[cc].v, false, bF.v, (short)0, acc, false, false);
#endif
            }
            #pragma unroll
            for (int v = 0; v < 8; ++v)
                mdnZ[(mBase + v) * 32 + colBase + nCol] = acc[v];
        }
        __syncthreads();

        // ---- phase 5: softmax / nnelu, emit mdn, keep as prev ----------
        for (int i = tid; i < 400; i += NTHREADS) {
            int m = i / 25, j = i % 25;
            float val;
            if (j < 8) {
                float mx = mdnZ[m * 32 + 0];
                #pragma unroll
                for (int k = 1; k < 8; ++k) mx = fmaxf(mx, mdnZ[m * 32 + k]);
                float sum = 0.f;
                #pragma unroll
                for (int k = 0; k < 8; ++k) sum += __expf(mdnZ[m * 32 + k] - mx);
                val = __expf(mdnZ[m * 32 + j] - mx) / sum;
            } else if (j < 16) {
                val = mdnZ[m * 32 + j];
            } else if (j < 24) {
                float zz = mdnZ[m * 32 + j];
                val = (zz > 0.f) ? (1.f + zz) : __expf(zz);   // 1 + elu
            } else {
                val = cmbLds[m];
            }
            prevLds[m * 25 + j] = val;
            out[(size_t)(baseRow + m) * (T_STEPS * 25) + t * 25 + j] = val;
        }
        __syncthreads();
    }
}

extern "C" void kernel_launch(void* const* d_in, const int* in_sizes, int n_in,
                              void* d_out, int out_size, void* d_ws, size_t ws_size,
                              hipStream_t stream) {
    // dict order: x, kernel, recurrent_kernel, bias, mlp_w1, mlp_b1, mlp_w2,
    //             mlp_b2, wa, ba, wm, bm, ws, bs
    const float* x    = (const float*)d_in[0];
    const float* kern = (const float*)d_in[1];
    const float* rk   = (const float*)d_in[2];
    const float* bias = (const float*)d_in[3];
    const float* w1   = (const float*)d_in[4];
    const float* b1   = (const float*)d_in[5];
    const float* w2   = (const float*)d_in[6];
    const float* b2   = (const float*)d_in[7];
    const float* wa   = (const float*)d_in[8];
    const float* ba   = (const float*)d_in[9];
    const float* wm   = (const float*)d_in[10];
    const float* bm   = (const float*)d_in[11];
    const float* wsig = (const float*)d_in[12];
    const float* bs   = (const float*)d_in[13];

    unsigned short* packed = (unsigned short*)d_ws;   // 77*11*512*2 = 867,328 B

    pack_weights<<<dim3(77 * 11), 64, 0, stream>>>(kern, rk, wa, wm, wsig, packed);
    dilstm_main<<<dim3(128), NTHREADS, 0, stream>>>(x, bias, w1, b1, w2, b2,
                                                    ba, bm, bs,
                                                    packed, (float*)d_out);
}
